// PointerSelectorV4_6_46780783788510
// MI455X (gfx1250) — matrix operand order repaired and checked by reading, not compiled, not run on hardware
//
#include <hip/hip_runtime.h>

typedef unsigned short u16;
typedef __attribute__((ext_vector_type(16))) __bf16 v16bf;
typedef __attribute__((ext_vector_type(8)))  float  v8f;

struct __attribute__((aligned(16))) U128 { unsigned x, y, z, w; };

union ABf {
  v16bf  v;
  __bf16 h[16];
  u16    u[16];
  U128   q[2];
};

// ---------------- helpers ----------------

// Let the compiler pick the HW conversion (v_cvt_pk_bf16_f32-class) instead of
// manual RNE bit math (which cost ~3 VALU/element in the GEMM inner loop).
__device__ __forceinline__ u16 f2bf(float f) {
  __bf16 h = (__bf16)f;
  return __builtin_bit_cast(u16, h);
}

// B-matrix fragment (32x16, K x N): lane n=lane&15 is column, half=lane>>4 picks K
// [k0+8*half..+7] and [k0+16+8*half..+7] -> two 16B runs from weight row n.
__device__ __forceinline__ v16bf load_b_frag(const u16* wrow, int k0, int half) {
  ABf f;
  f.q[0] = *(const U128*)(wrow + k0 + 8 * half);
  f.q[1] = *(const U128*)(wrow + k0 + 16 + 8 * half);
  return f.v;
}

// A-matrix fragment (16x32, M x K) from a bf16 row: lane m=lane&15 is the row,
// K pattern per ISA: lanes 0-15 hold K=[k0..k0+7] and [k0+16..k0+23];
// lanes 16-31 hold K=[k0+8..k0+15] and [k0+24..k0+31]. Two 16B runs.
__device__ __forceinline__ v16bf load_a_frag_lds(const u16* row, int k0, int half) {
  ABf f;
  f.q[0] = *(const U128*)(row + k0 + 8 * half);
  f.q[1] = *(const U128*)(row + k0 + 16 + 8 * half);
  return f.v;
}

// Same A layout but sourced from fp32 global memory with inline bf16 convert.
__device__ __forceinline__ v16bf load_a_frag_f32(const float* row, int k0, int half) {
  ABf f;
#pragma unroll
  for (int e = 0; e < 16; ++e) {
    int k = k0 + e + (e < 8 ? 0 : 8) + 8 * half;
    f.h[e] = (__bf16)row[k];
  }
  return f.v;
}

__device__ __forceinline__ v8f wmma_bf16(v16bf a, v16bf b, v8f c) {
  return __builtin_amdgcn_wmma_f32_16x16x32_bf16(false, a, false, b, (short)0, c,
                                                 false, false);
}

// ---------------- prep kernels ----------------

__global__ void cvt_bf16(const float* __restrict__ s, u16* __restrict__ d, int n) {
  int i = blockIdx.x * blockDim.x + threadIdx.x;
  if (i < n) d[i] = f2bf(s[i]);
}

__global__ void norm_protos(const float* __restrict__ p, float* __restrict__ d) {
  __shared__ float inv[16];
  int t = threadIdx.x;
  if (t < 16) {
    const float* r = p + t * 256;
    float s = 0.f;
    for (int j = 0; j < 256; ++j) s += r[j] * r[j];
    inv[t] = 1.f / fmaxf(sqrtf(s), 1e-12f);
  }
  __syncthreads();
  for (int i = t; i < 16 * 256; i += 256) d[i] = p[i] * inv[i >> 8];
}

// ---------------- big GEMM: C[M,256] = A_f32[M,Kd] @ W_bf16[256,Kd]^T ----------------
// 8 waves/block as a 4x2 wave grid, each wave owns a 32x32 tile (2x2 WMMA tiles).
// Block tile = 128(M) x 64(N). Grid = (M/128, 256/64).

__global__ void __launch_bounds__(256)
gemm_nt_bf16(const float* __restrict__ A, const u16* __restrict__ W, int Kd,
             u16* __restrict__ Cb, float* __restrict__ Cf) {
  const int lane = threadIdx.x & 31;
  const int wv   = threadIdx.x >> 5;
  const int wm = wv >> 1, wn = wv & 1;
  const int half = lane >> 4, l16 = lane & 15;
  const long row0 = (long)blockIdx.x * 128 + wm * 32;
  const int  col0 = blockIdx.y * 64 + wn * 32;

  v8f c00 = {}, c01 = {}, c10 = {}, c11 = {};
  for (int k0 = 0; k0 < Kd; k0 += 32) {
    v16bf a0 = load_a_frag_f32(A + (row0 + l16) * Kd, k0, half);
    v16bf a1 = load_a_frag_f32(A + (row0 + 16 + l16) * Kd, k0, half);
    v16bf b0 = load_b_frag(W + (size_t)(col0 + l16) * Kd, k0, half);
    v16bf b1 = load_b_frag(W + (size_t)(col0 + 16 + l16) * Kd, k0, half);
    c00 = wmma_bf16(a0, b0, c00);
    c01 = wmma_bf16(a0, b1, c01);
    c10 = wmma_bf16(a1, b0, c10);
    c11 = wmma_bf16(a1, b1, c11);
  }
#pragma unroll
  for (int r = 0; r < 8; ++r) {
    long m0 = row0 + r + 8 * half;
    long m1 = m0 + 16;
    int  n0 = col0 + l16, n1 = n0 + 16;
    if (Cb) {
      Cb[m0 * 256 + n0] = f2bf(c00[r]);
      Cb[m0 * 256 + n1] = f2bf(c01[r]);
      Cb[m1 * 256 + n0] = f2bf(c10[r]);
      Cb[m1 * 256 + n1] = f2bf(c11[r]);
    }
    if (Cf) {
      Cf[m0 * 256 + n0] = c00[r];
      Cf[m0 * 256 + n1] = c01[r];
      Cf[m1 * 256 + n0] = c10[r];
      Cf[m1 * 256 + n1] = c11[r];
    }
  }
}

// ---------------- fused per-row kernel ----------------
// One block (256 threads = 8 wave32) per batch row b. LDS holds the 32x256
// candidate tile (bf16) plus k/v/cn (f32) and all small per-row state.

#define HH 256
#define KC 32
#define TT 16
#define SS 2

__global__ void __launch_bounds__(256)
row_kernel(const u16* __restrict__ crb, const float* __restrict__ qr,
           const u16* __restrict__ Wkb, const u16* __restrict__ Wvb,
           const u16* __restrict__ Wcb, const float* __restrict__ protoN,
           const float* __restrict__ Wq_a, const float* __restrict__ bq_a,
           const float* __restrict__ bk_a, const float* __restrict__ bv_a,
           const float* __restrict__ Wo_a, const float* __restrict__ bo_a,
           const float* __restrict__ ln_w, const float* __restrict__ ln_b,
           const float* __restrict__ Wq,
           const float* __restrict__ W_ih, const float* __restrict__ W_hh,
           const float* __restrict__ b_ih, const float* __restrict__ b_hh,
           const float* __restrict__ step_emb,
           const float* __restrict__ Wqt, const float* __restrict__ bqt,
           const float* __restrict__ cover_lambda,
           float* __restrict__ out_logits, float* __restrict__ out_preds) {
  const int b = blockIdx.x;
  const int t = threadIdx.x;
  const int lane = t & 31;
  const int wv = t >> 5;
  const int half = lane >> 4, l16 = lane & 15;

  extern __shared__ char smem[];
  u16*   crs   = (u16*)smem;                       // 32x256 bf16    (16 KB)
  float* ks    = (float*)(smem + 16384);           // 32x256 f32     (32 KB)
  float* vs    = ks + KC * HH;                     // 32x256 f32
  float* cs    = vs + KC * HH;                     // 32x256 f32 (cn)
  float* qrs   = cs + KC * HH;                     // 256
  float* qs    = qrs + HH;                         // 256
  float* ctxs  = qs + HH;                          // 256
  float* xs    = ctxs + HH;                        // 256
  float* qes   = xs + HH;                          // 256
  float* hs    = qes + HH;                         // 256
  float* hsteps= hs + HH;                          // 256
  float* tmps  = hsteps + HH;                      // 256
  float* gis   = tmps + HH;                        // 768
  float* ghs   = gis + 3 * HH;                     // 768
  float* attns = ghs + 3 * HH;                     // 32
  float* tps   = attns + KC;                       // 32x16
  float* needs = tps + KC * TT;                    // 16
  float* covs  = needs + TT;                       // 16
  float* scs   = covs + TT;                        // 32 (norms, then sel mask)
  float* reds  = scs + KC;                         // 8 scratch scalars

  // stage candidate tile + query row
  {
    const U128* s4 = (const U128*)(crb + (size_t)b * KC * HH);
    U128* d4 = (U128*)crs;
    for (int i = t; i < KC * HH / 8; i += 256) d4[i] = s4[i];
  }
  qrs[t] = qr[(size_t)b * HH + t];
  __syncthreads();

  // k / v / cn GEMMs: 3 matrices x (2 M-tiles x 16 N-tiles) = 96 WMMA tiles,
  // 12 per wave, 8 WMMAs each over K=256.
  for (int tile = wv; tile < 96; tile += 8) {
    int mat = tile / 32;
    int rem = tile % 32;
    int mt = rem >> 4, nt = rem & 15;
    const u16*   Wb   = (mat == 0) ? Wkb : (mat == 1) ? Wvb : Wcb;
    const float* bias = (mat == 0) ? bk_a : (mat == 1) ? bv_a : nullptr;
    float*       Out  = (mat == 0) ? ks : (mat == 1) ? vs : cs;
    v8f c = {};
    for (int k0 = 0; k0 < HH; k0 += 32) {
      v16bf a = load_a_frag_lds(crs + (mt * 16 + l16) * HH, k0, half);
      v16bf bb = load_b_frag(Wb + (size_t)(nt * 16 + l16) * HH, k0, half);
      c = wmma_bf16(a, bb, c);
    }
    int n = nt * 16 + l16;
    float bval = bias ? bias[n] : 0.f;
#pragma unroll
    for (int r = 0; r < 8; ++r) {
      int m = mt * 16 + r + 8 * half;
      Out[m * HH + n] = c[r] + bval;
    }
  }
  __syncthreads();

  // q = qr @ Wq_a^T + bq_a
  {
    float acc = bq_a[t];
    const float* wr = Wq_a + t * HH;
    for (int j = 0; j < HH; ++j) acc = fmaf(qrs[j], wr[j], acc);
    qs[t] = acc;
  }
  __syncthreads();

  // attention logits over 32 keys, scale 1/sqrt(256)
  if (t < KC) {
    float acc = 0.f;
    const float* kr = ks + t * HH;
    for (int j = 0; j < HH; ++j) acc = fmaf(qs[j], kr[j], acc);
    attns[t] = acc * 0.0625f;
  }
  __syncthreads();
  if (t == 0) {
    float mx = attns[0];
    for (int i = 1; i < KC; ++i) mx = fmaxf(mx, attns[i]);
    float s = 0.f;
    for (int i = 0; i < KC; ++i) { float e = __expf(attns[i] - mx); attns[i] = e; s += e; }
    float inv = 1.f / s;
    for (int i = 0; i < KC; ++i) attns[i] *= inv;
  }
  __syncthreads();

  // ctx = attn @ v
  {
    float acc = 0.f;
    for (int k = 0; k < KC; ++k) acc = fmaf(attns[k], vs[k * HH + t], acc);
    ctxs[t] = acc;
  }
  __syncthreads();

  // out-proj + residual
  {
    float acc = bo_a[t];
    const float* wr = Wo_a + t * HH;
    for (int j = 0; j < HH; ++j) acc = fmaf(ctxs[j], wr[j], acc);
    xs[t] = acc + qrs[t];
  }
  __syncthreads();

  // LayerNorm
  if (t == 0) {
    float mu = 0.f;
    for (int j = 0; j < HH; ++j) mu += xs[j];
    mu *= (1.f / HH);
    float var = 0.f;
    for (int j = 0; j < HH; ++j) { float d = xs[j] - mu; var += d * d; }
    var *= (1.f / HH);
    reds[0] = mu;
    reds[1] = rsqrtf(var + 1e-5f);
  }
  __syncthreads();
  qes[t] = (xs[t] - reds[0]) * reds[1] * ln_w[t] + ln_b[t];
  __syncthreads();

  // h = l2n(qe @ Wq^T)
  {
    float acc = 0.f;
    const float* wr = Wq + t * HH;
    for (int j = 0; j < HH; ++j) acc = fmaf(qes[j], wr[j], acc);
    tmps[t] = acc;
  }
  __syncthreads();
  if (t == 0) {
    float s = 0.f;
    for (int j = 0; j < HH; ++j) s += tmps[j] * tmps[j];
    reds[0] = 1.f / fmaxf(sqrtf(s), 1e-12f);
  }
  __syncthreads();
  hs[t] = tmps[t] * reds[0];

  // normalize cn rows
  if (t < KC) {
    float s = 0.f;
    const float* cr = cs + t * HH;
    for (int j = 0; j < HH; ++j) s += cr[j] * cr[j];
    scs[t] = 1.f / fmaxf(sqrtf(s), 1e-12f);
  }
  __syncthreads();
  for (int i = t; i < KC * HH; i += 256) cs[i] *= scs[i >> 8];
  __syncthreads();

  // topic logits then per-candidate softmax over T
  for (int e = t; e < KC * TT; e += 256) {
    int k = e >> 4, tp = e & 15;
    float acc = 0.f;
    const float* cr = cs + k * HH;
    const float* pr = protoN + tp * HH;
    for (int j = 0; j < HH; ++j) acc = fmaf(cr[j], pr[j], acc);
    tps[e] = acc;
  }
  // need logits
  if (t < TT) {
    float acc = bqt[t];
    const float* wr = Wqt + t * HH;
    for (int j = 0; j < HH; ++j) acc = fmaf(hs[j], wr[j], acc);
    needs[t] = acc;
    covs[t] = 0.f;
  }
  __syncthreads();
  if (t < KC) {
    float* row = tps + t * TT;
    float mx = row[0];
    for (int i = 1; i < TT; ++i) mx = fmaxf(mx, row[i]);
    float s = 0.f;
    for (int i = 0; i < TT; ++i) { float e = __expf(row[i] - mx); row[i] = e; s += e; }
    float inv = 1.f / s;
    for (int i = 0; i < TT; ++i) row[i] *= inv;
  }
  if (t == 0) {
    float mx = needs[0];
    for (int i = 1; i < TT; ++i) mx = fmaxf(mx, needs[i]);
    float s = 0.f;
    for (int i = 0; i < TT; ++i) { float e = __expf(needs[i] - mx); needs[i] = e; s += e; }
    float inv = 1.f / s;
    for (int i = 0; i < TT; ++i) needs[i] *= inv;
    float cl = cover_lambda[0];
    reds[2] = (cl > 20.f) ? cl : log1pf(__expf(cl));  // softplus(lam)
  }
  if (t < KC) scs[t] = 0.f;  // selected mask
  __syncthreads();

  // selection loop
  for (int step = 0; step < SS; ++step) {
    const float* se = step_emb + (step < 8 ? step : 7) * HH;
    tmps[t] = hs[t] + se[t];
    __syncthreads();
    if (t == 0) {
      float s = 0.f;
      for (int j = 0; j < HH; ++j) s += tmps[j] * tmps[j];
      reds[0] = 1.f / fmaxf(sqrtf(s), 1e-12f);
    }
    __syncthreads();
    hsteps[t] = tmps[t] * reds[0];
    __syncthreads();

    if (t < KC) {
      float acc = 0.f;
      const float* cr = cs + t * HH;
      for (int j = 0; j < HH; ++j) acc = fmaf(hsteps[j], cr[j], acc);
      float sc = acc * 10.f;  // 1 / TEMP
      if (step > 0) {
        float g = 0.f;
        const float* tp = tps + t * TT;
        for (int i = 0; i < TT; ++i) {
          float un = fminf(fmaxf(1.f - covs[i], 0.f), 1.f);
          g = fmaf(needs[i] * un, tp[i], g);
        }
        sc += reds[2] * g;
      }
      if (scs[t] != 0.f) sc = -100.f;
      attns[t] = sc;
      out_logits[((size_t)b * SS + step) * KC + t] = sc;
    }
    __syncthreads();
    if (t == 0) {
      int idx = 0;
      float best = attns[0];
      for (int i = 1; i < KC; ++i)
        if (attns[i] > best) { best = attns[i]; idx = i; }
      reds[3] = (float)idx;
      out_preds[(size_t)b * SS + step] = (float)idx;
      scs[idx] = 1.f;
    }
    __syncthreads();
    int idx = (int)reds[3];
    if (t < TT) covs[t] = fminf(covs[t] + tps[idx * TT + t], 1.f);

    // GRU update (768 outputs, 3 per thread)
    const float* ch = cs + idx * HH;
    for (int o = t; o < 3 * HH; o += 256) {
      float gi = b_ih[o], gh = b_hh[o];
      const float* wi = W_ih + o * HH;
      const float* wh = W_hh + o * HH;
      for (int j = 0; j < HH; ++j) {
        gi = fmaf(ch[j], wi[j], gi);
        gh = fmaf(hs[j], wh[j], gh);
      }
      gis[o] = gi;
      ghs[o] = gh;
    }
    __syncthreads();
    {
      float r = 1.f / (1.f + __expf(-(gis[t] + ghs[t])));
      float z = 1.f / (1.f + __expf(-(gis[HH + t] + ghs[HH + t])));
      float n = tanhf(gis[2 * HH + t] + r * ghs[2 * HH + t]);
      tmps[t] = (1.f - z) * n + z * hs[t];
    }
    __syncthreads();
    if (t == 0) {
      float s = 0.f;
      for (int j = 0; j < HH; ++j) s += tmps[j] * tmps[j];
      reds[0] = 1.f / fmaxf(sqrtf(s), 1e-12f);
    }
    __syncthreads();
    hs[t] = tmps[t] * reds[0];
    __syncthreads();
  }
}

// ---------------- launch ----------------

extern "C" void kernel_launch(void* const* d_in, const int* in_sizes, int n_in,
                              void* d_out, int out_size, void* d_ws, size_t ws_size,
                              hipStream_t stream) {
  (void)in_sizes; (void)n_in; (void)out_size; (void)ws_size;
  const int B = 8192, K = 32, D = 768, H = 256;

  const float* query = (const float*)d_in[0];
  const float* cand  = (const float*)d_in[1];
  const float* W_in  = (const float*)d_in[2];
  const float* Wq_a  = (const float*)d_in[3];
  const float* Wk_a  = (const float*)d_in[4];
  const float* Wv_a  = (const float*)d_in[5];
  const float* bq_a  = (const float*)d_in[6];
  const float* bk_a  = (const float*)d_in[7];
  const float* bv_a  = (const float*)d_in[8];
  const float* Wo_a  = (const float*)d_in[9];
  const float* bo_a  = (const float*)d_in[10];
  const float* ln_w  = (const float*)d_in[11];
  const float* ln_b  = (const float*)d_in[12];
  const float* Wqp   = (const float*)d_in[13];
  const float* Wcp   = (const float*)d_in[14];
  const float* W_ih  = (const float*)d_in[15];
  const float* W_hh  = (const float*)d_in[16];
  const float* b_ih  = (const float*)d_in[17];
  const float* b_hh  = (const float*)d_in[18];
  const float* stepe = (const float*)d_in[19];
  const float* protos= (const float*)d_in[20];
  const float* Wqt   = (const float*)d_in[21];
  const float* bqt   = (const float*)d_in[22];
  const float* clam  = (const float*)d_in[23];

  char* ws = (char*)d_ws;
  size_t off = 0;
  u16* Winb = (u16*)(ws + off); off += (size_t)H * D * 2;       // 393216
  u16* Wkb  = (u16*)(ws + off); off += (size_t)H * H * 2;       // 131072
  u16* Wvb  = (u16*)(ws + off); off += (size_t)H * H * 2;
  u16* Wcb  = (u16*)(ws + off); off += (size_t)H * H * 2;
  float* protoN = (float*)(ws + off); off += (size_t)16 * H * 4;
  float* qr = (float*)(ws + off); off += (size_t)B * H * 4;
  u16* crb  = (u16*)(ws + off); off += (size_t)B * K * H * 2;

  cvt_bf16<<<(H * D + 255) / 256, 256, 0, stream>>>(W_in, Winb, H * D);
  cvt_bf16<<<(H * H + 255) / 256, 256, 0, stream>>>(Wk_a, Wkb, H * H);
  cvt_bf16<<<(H * H + 255) / 256, 256, 0, stream>>>(Wv_a, Wvb, H * H);
  cvt_bf16<<<(H * H + 255) / 256, 256, 0, stream>>>(Wcp, Wcb, H * H);
  norm_protos<<<1, 256, 0, stream>>>(protos, protoN);

  // cr (bf16): [B*K, 768] x [256, 768]^T
  gemm_nt_bf16<<<dim3((B * K) / 128, 4), 256, 0, stream>>>(cand, Winb, D, crb, nullptr);
  // qr (f32): [B, 768] x [256, 768]^T
  gemm_nt_bf16<<<dim3(B / 128, 4), 256, 0, stream>>>(query, Winb, D, nullptr, qr);

  float* out_logits = (float*)d_out;
  float* out_preds  = out_logits + (size_t)B * 2 * 32;

  const size_t kRowSmem = 131488;  // crs(16K)+k/v/cn(96K)+per-row state
  row_kernel<<<B, 256, kRowSmem, stream>>>(
      crb, qr, Wkb, Wvb, Wcb, protoN, Wq_a, bq_a, bk_a, bv_a, Wo_a, bo_a,
      ln_w, ln_b, Wqp, W_ih, W_hh, b_ih, b_hh, stepe, Wqt, bqt, clam,
      out_logits, out_preds);
}
